// DualTransformerBlock_2972117369306
// MI455X (gfx1250) — matrix operand order repaired and checked by reading, not compiled, not run on hardware
//
#include <hip/hip_runtime.h>
#include <cstdint>
#include <cstddef>

// ---------------------------------------------------------------------------
// Types
// ---------------------------------------------------------------------------
typedef __bf16 bf16;
typedef __attribute__((ext_vector_type(16))) __bf16 v16bf;
typedef __attribute__((ext_vector_type(8)))  __bf16 v8bf;
typedef __attribute__((ext_vector_type(8)))  float  v8f;
typedef __attribute__((ext_vector_type(4)))  unsigned int u32x4;
typedef __attribute__((ext_vector_type(8)))  int i32x8;
typedef __attribute__((ext_vector_type(4)))  int i32x4;

#define TOKENS   4096
#define BATCH    16
#define CDIM     256
#define HIDDEN   1024
#define NHEADS   8
#define MROWS    (BATCH * TOKENS)   // 65536

// Tensor Data Mover availability (probe: builtin exists on both toolchains,
// arity differs: ROCm7.2/clang-22 = 5 args, amdgpu-toolchain/clang-23 = 6 args)
#ifdef __has_builtin
# if __has_builtin(__builtin_amdgcn_tensor_load_to_lds)
#  define HAVE_TDM 1
# endif
#endif
#ifndef HAVE_TDM
# define HAVE_TDM 0
#endif

// ---------------------------------------------------------------------------
// fp32 -> bf16 transposed weight conversion: out[n*K+k] = (bf16)in[k*N+n]
// ---------------------------------------------------------------------------
__global__ __launch_bounds__(256)
void f2bf_t_kernel(const float* __restrict__ in, bf16* __restrict__ out,
                   int K, int N) {
    int k = blockIdx.x * 16 + (threadIdx.x & 15);
    int n = blockIdx.y * 16 + (threadIdx.x >> 4);
    out[(size_t)n * K + k] = (bf16)in[(size_t)k * N + n];
}

// ---------------------------------------------------------------------------
// LayerNorm (row of 256) -> bf16.  One wave per row, 8 rows per block.
// ---------------------------------------------------------------------------
__global__ __launch_bounds__(256)
void ln_bf16_kernel(const float* __restrict__ x, const float* __restrict__ g,
                    const float* __restrict__ bvec, bf16* __restrict__ out) {
    int wave = threadIdx.x >> 5, lane = threadIdx.x & 31;
    size_t row = (size_t)blockIdx.x * 8 + wave;
    const float* xr = x + row * CDIM;
    float vals[8];
    float s = 0.f;
    #pragma unroll
    for (int j = 0; j < 8; j++) { vals[j] = xr[lane + j * 32]; s += vals[j]; }
    #pragma unroll
    for (int off = 16; off; off >>= 1) s += __shfl_xor(s, off, 32);
    float mean = s * (1.f / 256.f);
    float vs = 0.f;
    #pragma unroll
    for (int j = 0; j < 8; j++) { float d = vals[j] - mean; vs += d * d; }
    #pragma unroll
    for (int off = 16; off; off >>= 1) vs += __shfl_xor(vs, off, 32);
    float inv = rsqrtf(vs * (1.f / 256.f) + 1e-5f);
    bf16* orow = out + row * CDIM;
    #pragma unroll
    for (int j = 0; j < 8; j++) {
        int c = lane + j * 32;
        orow[c] = (bf16)((vals[j] - mean) * inv * g[c] + bvec[c]);
    }
}

// ---------------------------------------------------------------------------
// Fused LN1 + Haar wavelet mixing + residual:  add1 = x + wavelet(LN1(x))
// ---------------------------------------------------------------------------
__global__ __launch_bounds__(128)
void ln_wavelet_kernel(const float* __restrict__ x,
                       const float* __restrict__ g, const float* __restrict__ bvec,
                       const float* __restrict__ alpha, const float* __restrict__ beta,
                       const float* __restrict__ w1, const float* __restrict__ b1,
                       const float* __restrict__ w2, const float* __restrict__ b2,
                       float* __restrict__ add1) {
    __shared__ float lnv[4][CDIM];
    int blk = blockIdx.x;
    int b  = blk >> 10;
    int h2 = (blk >> 5) & 31;
    int w2_ = blk & 31;
    int rows[4];
    rows[0] = (2 * h2)     * 64 + 2 * w2_;
    rows[1] = (2 * h2)     * 64 + 2 * w2_ + 1;
    rows[2] = (2 * h2 + 1) * 64 + 2 * w2_;
    rows[3] = (2 * h2 + 1) * 64 + 2 * w2_ + 1;
    int wave = threadIdx.x >> 5, lane = threadIdx.x & 31;

    const float* xr = x + ((size_t)b * TOKENS + rows[wave]) * CDIM;
    float vals[8];
    float s = 0.f;
    #pragma unroll
    for (int j = 0; j < 8; j++) { vals[j] = xr[lane + j * 32]; s += vals[j]; }
    #pragma unroll
    for (int off = 16; off; off >>= 1) s += __shfl_xor(s, off, 32);
    float mean = s * (1.f / 256.f);
    float vs = 0.f;
    #pragma unroll
    for (int j = 0; j < 8; j++) { float d = vals[j] - mean; vs += d * d; }
    #pragma unroll
    for (int off = 16; off; off >>= 1) vs += __shfl_xor(vs, off, 32);
    float inv = rsqrtf(vs * (1.f / 256.f) + 1e-5f);
    #pragma unroll
    for (int j = 0; j < 8; j++) {
        int c = lane + j * 32;
        lnv[wave][c] = (vals[j] - mean) * inv * g[c] + bvec[c];
    }
    __syncthreads();

    float W1 = w1[0], B1 = b1[0], W2 = w2[0], B2 = b2[0], AL = alpha[0], BE = beta[0];
    #pragma unroll
    for (int i = 0; i < 2; i++) {
        int c = threadIdx.x + i * 128;
        float a  = lnv[0][c], bb = lnv[1][c], cc = lnv[2][c], dd = lnv[3][c];
        float cA = (a + bb + cc + dd) * 0.5f;
        float cH = (a - bb + cc - dd) * 0.5f;
        float cV = (a + bb - cc - dd) * 0.5f;
        float cD = (a - bb - cc + dd) * 0.5f;
        cA = W1 * cA + B1;
        cH = W2 * cH + B2;
        cV = W2 * cV + B2;
        cD = W2 * cD + B2;
        float comb = AL * cA + BE * (cH + cV + cD);
        float aa2 = (comb + cH + cV + cD) * 0.5f;
        float bb2 = (comb - cH + cV - cD) * 0.5f;
        float cc2 = (comb + cH - cV - cD) * 0.5f;
        float dd2 = (comb - cH - cV + cD) * 0.5f;
        size_t base = (size_t)b * TOKENS * CDIM;
        size_t p0 = base + (size_t)rows[0] * CDIM + c;
        size_t p1 = base + (size_t)rows[1] * CDIM + c;
        size_t p2 = base + (size_t)rows[2] * CDIM + c;
        size_t p3 = base + (size_t)rows[3] * CDIM + c;
        add1[p0] = x[p0] + aa2;
        add1[p1] = x[p1] + bb2;
        add1[p2] = x[p2] + cc2;
        add1[p3] = x[p3] + dd2;
    }
}

// ---------------------------------------------------------------------------
// TDM helper: async-load one 2D tile (32 K-elems x 128 rows, bf16) into LDS.
// Tensor: row stride = strideK elements.  LDS: padded to 40-element rows
// (pad_interval = 16 DWORDs, pad_amount = 4 DWORDs -> 64B data + 16B pad).
// ---------------------------------------------------------------------------
#if HAVE_TDM
__device__ __forceinline__ void tdm_load_tile(bf16* ldsp, const bf16* g, int strideK) {
    uint32_t laddr = (uint32_t)(uintptr_t)(void*)ldsp;  // low 32 bits = LDS offset
    uint64_t ga = (uint64_t)(uintptr_t)g;
    u32x4 g0;
    g0[0] = 1u;                                              // count=1 (valid D#)
    g0[1] = laddr;                                           // lds_addr
    g0[2] = (uint32_t)ga;                                    // global_addr[31:0]
    g0[3] = (uint32_t)((ga >> 32) & 0x01FFFFFFu) | (2u << 30); // addr[56:32] | type=2
    i32x8 g1;
    g1[0] = (1 << 16) | (1 << 20) | (3 << 22) | (3 << 25);   // 2B elems, pad en, 16dw/4dw
    g1[1] = (int)(32u << 16);        // tensor_dim0 = 32 (lo16 @ bits 63:48)
    g1[2] = (int)(128u << 16);       // tensor_dim0 hi=0 | tensor_dim1 = 128 (lo16)
    g1[3] = (int)(32u << 16);        // tensor_dim1 hi=0 | tile_dim0 = 32
    g1[4] = 128;                     // tile_dim1 = 128, tile_dim2 = 0
    g1[5] = strideK;                 // tensor_dim0_stride lo32
    g1[6] = 0;                       // stride hi | dim1_stride lo (unused for 2D)
    g1[7] = 0;
    i32x4 z4 = {0, 0, 0, 0};
#if __clang_major__ >= 23
    i32x8 z8 = {0, 0, 0, 0, 0, 0, 0, 0};
    __builtin_amdgcn_tensor_load_to_lds(g0, g1, z4, z4, z8, 0);
#else
    __builtin_amdgcn_tensor_load_to_lds(g0, g1, z4, z4, 0);
#endif
}
#endif

// ---------------------------------------------------------------------------
// bf16 WMMA GEMM: C[M,N] = A[M,K] * Bt[N,K]^T (+bias) (+residual)
//   EPI 0: store bf16 (no bias)  EPI 1: +bias, bf16  EPI 2: +bias+resid, fp32
// 256 threads = 8 waves; 128x128 tile; BK=32; wave = 32(M) x 64(N).
// TDM path: double-buffered LDS, tiles streamed by the Tensor Data Mover.
// ---------------------------------------------------------------------------
#define BM 128
#define BN 128
#define BK 32
#define LSTR 40   // LDS row stride in elements (80B, 16B aligned, conflict-free)
#define TILE_ELEMS (BM * LSTR)

template <int EPI>
__global__ __launch_bounds__(256)
void gemm_bf16_kernel(const bf16* __restrict__ A, const bf16* __restrict__ Bt,
                      const float* __restrict__ bias, const float* __restrict__ resid,
                      float* __restrict__ outF, bf16* __restrict__ outB,
                      int M, int N, int K) {
    const int t    = threadIdx.x;
    const int wave = t >> 5;
    const int lane = t & 31;
    const int wm   = wave & 3;    // M sub-tile *32
    const int wn   = wave >> 2;   // N sub-tile *64
    const int rowBase = blockIdx.y * BM;
    const int colBase = blockIdx.x * BN;

    v8f acc[2][4];
    #pragma unroll
    for (int i = 0; i < 2; i++)
        #pragma unroll
        for (int j = 0; j < 4; j++) acc[i][j] = v8f{};

#if HAVE_TDM
    __shared__ bf16 smem[4 * TILE_ELEMS];   // [buf0 A | buf0 B | buf1 A | buf1 B]
    const int wv0 = __builtin_amdgcn_readfirstlane(wave);  // force scalar branch:
                                                           // tensor ops ignore EXEC
    if (wv0 == 0) {
        tdm_load_tile(smem,              A  + (size_t)rowBase * K, K);
        tdm_load_tile(smem + TILE_ELEMS, Bt + (size_t)colBase * K, K);
        __builtin_amdgcn_s_wait_tensorcnt(0);
    }
    __syncthreads();

    const int nsteps = K / BK;
    for (int step = 0; step < nsteps; ++step) {
        const int curOff  = (step & 1) * (2 * TILE_ELEMS);
        const int nextOff = curOff ^ (2 * TILE_ELEMS);
        const bf16* Acur = smem + curOff;
        const bf16* Bcur = smem + curOff + TILE_ELEMS;
        if (wv0 == 0 && step + 1 < nsteps) {
            const int k0n = (step + 1) * BK;
            tdm_load_tile(smem + nextOff,              A  + (size_t)rowBase * K + k0n, K);
            tdm_load_tile(smem + nextOff + TILE_ELEMS, Bt + (size_t)colBase * K + k0n, K);
        }
#else
    __shared__ bf16 smem[2 * TILE_ELEMS];
    const int arow = t >> 2;          // 0..63
    const int acol = (t & 3) * 8;     // 0,8,16,24

    for (int k0 = 0; k0 < K; k0 += BK) {
        #pragma unroll
        for (int p = 0; p < 2; p++) {
            int r = p * 64 + arow;
            *(uint4*)(&smem[r * LSTR + acol]) =
                *(const uint4*)(A  + (size_t)(rowBase + r) * K + k0 + acol);
            *(uint4*)(&smem[TILE_ELEMS + r * LSTR + acol]) =
                *(const uint4*)(Bt + (size_t)(colBase + r) * K + k0 + acol);
        }
        __syncthreads();
        const bf16* Acur = smem;
        const bf16* Bcur = smem + TILE_ELEMS;
#endif
        // A fragments (16x32): lane<16 holds K{0..7,16..23}, lane>=16 K{8..15,24..31}
        v16bf af[2];
        #pragma unroll
        for (int mf = 0; mf < 2; mf++) {
            int r  = wm * 32 + mf * 16 + (lane & 15);
            int kg = (lane >> 4) * 8;
            const bf16* ap = &Acur[r * LSTR + kg];
            v8bf lo = *(const v8bf*)ap;
            v8bf hi = *(const v8bf*)(ap + 16);
            v16bf v;
            #pragma unroll
            for (int j = 0; j < 8; j++) { v[j] = lo[j]; v[8 + j] = hi[j]; }
            af[mf] = v;
        }
        // B fragments (32x16): n = lane%16, lane<16 holds K{0..7,16..23}, lane>=16 K{8..15,24..31}
        #pragma unroll
        for (int nf = 0; nf < 4; nf++) {
            int n  = wn * 64 + nf * 16 + (lane & 15);
            int ks = (lane >> 4) * 8;
            const bf16* bp = &Bcur[n * LSTR + ks];
            v8bf lo = *(const v8bf*)bp;
            v8bf hi = *(const v8bf*)(bp + 16);
            v16bf bv;
            #pragma unroll
            for (int j = 0; j < 8; j++) { bv[j] = lo[j]; bv[8 + j] = hi[j]; }
            #pragma unroll
            for (int mf = 0; mf < 2; mf++) {
                acc[mf][nf] = __builtin_amdgcn_wmma_f32_16x16x32_bf16(
                    false, af[mf], false, bv, (short)0, acc[mf][nf], false, false);
            }
        }
#if HAVE_TDM
        if (wv0 == 0) __builtin_amdgcn_s_wait_tensorcnt(0);
        __syncthreads();
    }
#else
        __syncthreads();
    }
#endif

    // Epilogue. C/D layout: n = lane%16; lane<16 -> M = j; lane>=16 -> M = 8+j
    #pragma unroll
    for (int mf = 0; mf < 2; mf++) {
        #pragma unroll
        for (int nf = 0; nf < 4; nf++) {
            int col   = colBase + wn * 64 + nf * 16 + (lane & 15);
            int rbase = rowBase + wm * 32 + mf * 16 + ((lane >> 4) ? 8 : 0);
            float bv = 0.f;
            if (EPI >= 1) bv = bias[col];
            #pragma unroll
            for (int j = 0; j < 8; j++) {
                size_t idx = (size_t)(rbase + j) * N + col;
                float v = acc[mf][nf][j] + bv;
                if (EPI == 2) outF[idx] = v + resid[idx];
                else          outB[idx] = (bf16)v;
            }
        }
    }
}

// ---------------------------------------------------------------------------
// Depthwise 3x3 conv (pad 1) + bias + exact GELU, (B,HID,64,64), bf16 in/out.
// ---------------------------------------------------------------------------
__global__ __launch_bounds__(256)
void dwconv_gelu_kernel(const bf16* __restrict__ h, const float* __restrict__ w,
                        const float* __restrict__ bias, bf16* __restrict__ out) {
    int pix = blockIdx.x;            // b*4096 + hh*64 + ww
    int b   = pix >> 12;
    int n   = pix & 4095;
    int hh  = n >> 6, ww = n & 63;
    #pragma unroll
    for (int i = 0; i < 4; i++) {
        int c = threadIdx.x + i * 256;
        float acc = bias[c];
        #pragma unroll
        for (int dy = -1; dy <= 1; dy++) {
            int y = hh + dy;
            if ((unsigned)y >= 64u) continue;
            #pragma unroll
            for (int dx = -1; dx <= 1; dx++) {
                int xw = ww + dx;
                if ((unsigned)xw >= 64u) continue;
                float wv = w[c * 9 + (dy + 1) * 3 + (dx + 1)];
                acc += wv * (float)h[(((size_t)b * TOKENS) + y * 64 + xw) * HIDDEN + c];
            }
        }
        float ge = 0.5f * acc * (1.f + erff(acc * 0.70710678118654752f));
        out[(size_t)pix * HIDDEN + c] = (bf16)ge;
    }
}

// ---------------------------------------------------------------------------
// Channel attention, stage 1: per (b,head) Gram matrix over N, L2 norms,
// temperature scale, softmax -> attn[b,h,32,32].
// ---------------------------------------------------------------------------
__global__ __launch_bounds__(256)
void chattn_qk_kernel(const bf16* __restrict__ qkv, const float* __restrict__ temp,
                      float* __restrict__ attnOut) {
    int b = blockIdx.x >> 3;
    int head = blockIdx.x & 7;
    __shared__ float qs[32][65];
    __shared__ float ks[32][65];
    __shared__ float red[64];
    __shared__ float Srow[32][33];

    int t  = threadIdx.x;
    int d  = t >> 3;          // 0..31
    int e0 = (t & 7) * 4;     // 0..28
    float G[4] = {0.f, 0.f, 0.f, 0.f};
    float nacc = 0.f;
    const size_t qbase = (size_t)b * TOKENS * 768 + head * 32;
    const size_t kbase = qbase + 256;

    int ld_d  = t & 31;
    int ld_n0 = t >> 5;       // 0..7

    for (int n0 = 0; n0 < TOKENS; n0 += 64) {
        #pragma unroll
        for (int j = 0; j < 8; j++) {
            int nn = n0 + ld_n0 + j * 8;
            qs[ld_d][ld_n0 + j * 8] = (float)qkv[qbase + (size_t)nn * 768 + ld_d];
            ks[ld_d][ld_n0 + j * 8] = (float)qkv[kbase + (size_t)nn * 768 + ld_d];
        }
        __syncthreads();
        #pragma unroll 4
        for (int j = 0; j < 64; j++) {
            float qv = qs[d][j];
            G[0] += qv * ks[e0 + 0][j];
            G[1] += qv * ks[e0 + 1][j];
            G[2] += qv * ks[e0 + 2][j];
            G[3] += qv * ks[e0 + 3][j];
        }
        if (t < 32) {
            #pragma unroll 4
            for (int j = 0; j < 64; j++) { float q2 = qs[t][j]; nacc += q2 * q2; }
        } else if (t < 64) {
            #pragma unroll 4
            for (int j = 0; j < 64; j++) { float k2 = ks[t - 32][j]; nacc += k2 * k2; }
        }
        __syncthreads();
    }
    if (t < 64) red[t] = sqrtf(nacc);
    __syncthreads();

    float tv = temp[head];
    float nq = fmaxf(red[d], 1e-12f);
    #pragma unroll
    for (int j = 0; j < 4; j++) {
        float nk = fmaxf(red[32 + e0 + j], 1e-12f);
        Srow[d][e0 + j] = G[j] / (nq * nk) * tv;
    }
    __syncthreads();
    if (t < 32) {
        float mx = -1e30f;
        #pragma unroll
        for (int e = 0; e < 32; e++) mx = fmaxf(mx, Srow[t][e]);
        float ex[32];
        float ssum = 0.f;
        #pragma unroll
        for (int e = 0; e < 32; e++) { ex[e] = __expf(Srow[t][e] - mx); ssum += ex[e]; }
        float inv = 1.f / ssum;
        float* dst = attnOut + (size_t)blockIdx.x * 1024 + t * 32;
        #pragma unroll
        for (int e = 0; e < 32; e++) dst[e] = ex[e] * inv;
    }
}

// ---------------------------------------------------------------------------
// Channel attention, stage 2: o[b,n,head*32+d] = sum_e attn[d][e] * v[e][n]
// ---------------------------------------------------------------------------
__global__ __launch_bounds__(256)
void chattn_av_kernel(const bf16* __restrict__ qkv, const float* __restrict__ attn,
                      bf16* __restrict__ obuf) {
    int bh = blockIdx.x;
    int b  = bh >> 3;
    int head = bh & 7;
    int n = blockIdx.y * 256 + threadIdx.x;
    __shared__ float As2[32][33];
    for (int i = threadIdx.x; i < 1024; i += 256)
        As2[i >> 5][i & 31] = attn[(size_t)bh * 1024 + i];
    __syncthreads();

    const size_t vbase = ((size_t)b * TOKENS + n) * 768 + 512 + head * 32;
    float v[32];
    #pragma unroll
    for (int e = 0; e < 32; e++) v[e] = (float)qkv[vbase + e];
    bf16* orow = obuf + ((size_t)b * TOKENS + n) * CDIM + head * 32;
    #pragma unroll
    for (int d = 0; d < 32; d++) {
        float acc = 0.f;
        #pragma unroll
        for (int e = 0; e < 32; e++) acc += As2[d][e] * v[e];
        orow[d] = (bf16)acc;
    }
}

// ---------------------------------------------------------------------------
// Host-side orchestration
// ---------------------------------------------------------------------------
extern "C" void kernel_launch(void* const* d_in, const int* in_sizes, int n_in,
                              void* d_out, int out_size, void* d_ws, size_t ws_size,
                              hipStream_t stream) {
    const float* x     = (const float*)d_in[0];
    const float* ln1_g = (const float*)d_in[1];
    const float* ln1_b = (const float*)d_in[2];
    const float* ln2_g = (const float*)d_in[3];
    const float* ln2_b = (const float*)d_in[4];
    const float* ln3_g = (const float*)d_in[5];
    const float* ln3_b = (const float*)d_in[6];
    const float* ln4_g = (const float*)d_in[7];
    const float* ln4_b = (const float*)d_in[8];
    const float* alpha = (const float*)d_in[9];
    const float* beta  = (const float*)d_in[10];
    const float* w1    = (const float*)d_in[11];
    const float* b1    = (const float*)d_in[12];
    const float* w2    = (const float*)d_in[13];
    const float* b2    = (const float*)d_in[14];
    const float* temp  = (const float*)d_in[15];
    const float* qkv_w = (const float*)d_in[16];
    const float* proj_w = (const float*)d_in[17];
    const float* proj_b = (const float*)d_in[18];
    const float* m1_fc1w = (const float*)d_in[19];
    const float* m1_fc1b = (const float*)d_in[20];
    const float* m1_dww  = (const float*)d_in[21];
    const float* m1_dwb  = (const float*)d_in[22];
    const float* m1_fc2w = (const float*)d_in[23];
    const float* m1_fc2b = (const float*)d_in[24];
    const float* m2_fc1w = (const float*)d_in[25];
    const float* m2_fc1b = (const float*)d_in[26];
    const float* m2_dww  = (const float*)d_in[27];
    const float* m2_dwb  = (const float*)d_in[28];
    const float* m2_fc2w = (const float*)d_in[29];
    const float* m2_fc2b = (const float*)d_in[30];
    float* out = (float*)d_out;

    // Workspace layout (bytes).  Transposed bf16 weights first.
    uint8_t* ws = (uint8_t*)d_ws;
    constexpr size_t OFF_WQKV  = 0;
    constexpr size_t OFF_WPROJ = OFF_WQKV  + (size_t)CDIM * 768 * 2;
    constexpr size_t OFF_WFC1A = OFF_WPROJ + (size_t)CDIM * CDIM * 2;
    constexpr size_t OFF_WFC2A = OFF_WFC1A + (size_t)CDIM * HIDDEN * 2;
    constexpr size_t OFF_WFC1B = OFF_WFC2A + (size_t)HIDDEN * CDIM * 2;
    constexpr size_t OFF_WFC2B = OFF_WFC1B + (size_t)CDIM * HIDDEN * 2;
    constexpr size_t OFF_ADD1  = OFF_WFC2B + (size_t)HIDDEN * CDIM * 2;
    constexpr size_t OFF_ADD2  = OFF_ADD1  + (size_t)MROWS * CDIM * 4;
    constexpr size_t OFF_ADD3  = OFF_ADD2  + (size_t)MROWS * CDIM * 4;
    constexpr size_t OFF_LN    = OFF_ADD3  + (size_t)MROWS * CDIM * 4;  // bf16 (reused)
    constexpr size_t OFF_QKV   = OFF_LN    + (size_t)MROWS * CDIM * 2;
    constexpr size_t OFF_ATTN  = OFF_QKV   + (size_t)MROWS * 768 * 2;
    constexpr size_t OFF_H     = OFF_ATTN  + (size_t)BATCH * NHEADS * 1024 * 4;
    constexpr size_t OFF_DW    = OFF_H     + (size_t)MROWS * HIDDEN * 2;

    bf16*  wqkv  = (bf16*)(ws + OFF_WQKV);
    bf16*  wproj = (bf16*)(ws + OFF_WPROJ);
    bf16*  wfc1a = (bf16*)(ws + OFF_WFC1A);
    bf16*  wfc2a = (bf16*)(ws + OFF_WFC2A);
    bf16*  wfc1b = (bf16*)(ws + OFF_WFC1B);
    bf16*  wfc2b = (bf16*)(ws + OFF_WFC2B);
    float* add1  = (float*)(ws + OFF_ADD1);
    float* add2  = (float*)(ws + OFF_ADD2);
    float* add3  = (float*)(ws + OFF_ADD3);
    bf16*  lnbuf = (bf16*)(ws + OFF_LN);
    bf16*  qkvb  = (bf16*)(ws + OFF_QKV);
    float* attnb = (float*)(ws + OFF_ATTN);
    bf16*  hbuf  = (bf16*)(ws + OFF_H);
    bf16*  dwbuf = (bf16*)(ws + OFF_DW);

    // Convert + transpose weights to bf16 [N][K]
    auto cvtT = [&](const float* src, bf16* dst, int K, int N) {
        f2bf_t_kernel<<<dim3(K / 16, N / 16), 256, 0, stream>>>(src, dst, K, N);
    };
    cvtT(qkv_w,   wqkv,  CDIM, 768);
    cvtT(proj_w,  wproj, CDIM, CDIM);
    cvtT(m1_fc1w, wfc1a, CDIM, HIDDEN);
    cvtT(m1_fc2w, wfc2a, HIDDEN, CDIM);
    cvtT(m2_fc1w, wfc1b, CDIM, HIDDEN);
    cvtT(m2_fc2w, wfc2b, HIDDEN, CDIM);

    const dim3 blk256(256);
    const dim3 gemmN256(CDIM / BN,    MROWS / BM);   // (2, 512)
    const dim3 gemmN768(768 / BN,     MROWS / BM);   // (6, 512)
    const dim3 gemmN1024(HIDDEN / BN, MROWS / BM);   // (8, 512)

    // --- Stage 1: add1 = x + wavelet(LN1(x))
    ln_wavelet_kernel<<<BATCH * 32 * 32, 128, 0, stream>>>(
        x, ln1_g, ln1_b, alpha, beta, w1, b1, w2, b2, add1);

    // --- Stage 2: mixffn #1 -> add2
    ln_bf16_kernel<<<MROWS / 8, blk256, 0, stream>>>(add1, ln2_g, ln2_b, lnbuf);
    gemm_bf16_kernel<1><<<gemmN1024, blk256, 0, stream>>>(
        lnbuf, wfc1a, m1_fc1b, nullptr, nullptr, hbuf, MROWS, HIDDEN, CDIM);
    dwconv_gelu_kernel<<<MROWS, blk256, 0, stream>>>(hbuf, m1_dww, m1_dwb, dwbuf);
    gemm_bf16_kernel<2><<<gemmN256, blk256, 0, stream>>>(
        dwbuf, wfc2a, m1_fc2b, add1, add2, nullptr, MROWS, CDIM, HIDDEN);

    // --- Stage 3: channel attention -> add3
    ln_bf16_kernel<<<MROWS / 8, blk256, 0, stream>>>(add2, ln3_g, ln3_b, lnbuf);
    gemm_bf16_kernel<0><<<gemmN768, blk256, 0, stream>>>(
        lnbuf, wqkv, nullptr, nullptr, nullptr, qkvb, MROWS, 768, CDIM);
    chattn_qk_kernel<<<BATCH * NHEADS, blk256, 0, stream>>>(qkvb, temp, attnb);
    chattn_av_kernel<<<dim3(BATCH * NHEADS, TOKENS / 256), blk256, 0, stream>>>(
        qkvb, attnb, lnbuf);   // lnbuf reused as o (bf16, M x C)
    gemm_bf16_kernel<2><<<gemmN256, blk256, 0, stream>>>(
        lnbuf, wproj, proj_b, add2, add3, nullptr, MROWS, CDIM, CDIM);

    // --- Stage 4: mixffn #2 -> out
    ln_bf16_kernel<<<MROWS / 8, blk256, 0, stream>>>(add3, ln4_g, ln4_b, lnbuf);
    gemm_bf16_kernel<1><<<gemmN1024, blk256, 0, stream>>>(
        lnbuf, wfc1b, m2_fc1b, nullptr, nullptr, hbuf, MROWS, HIDDEN, CDIM);
    dwconv_gelu_kernel<<<MROWS, blk256, 0, stream>>>(hbuf, m2_dww, m2_dwb, dwbuf);
    gemm_bf16_kernel<2><<<gemmN256, blk256, 0, stream>>>(
        dwbuf, wfc2b, m2_fc2b, add3, out, nullptr, MROWS, CDIM, HIDDEN);

    (void)in_sizes; (void)n_in; (void)out_size; (void)ws_size;
}
